// IntLoRA_SHIFT_81939386073209
// MI455X (gfx1250) — matrix operand order repaired and checked by reading, not compiled, not run on hardware
//
#include <hip/hip_runtime.h>
#include <hip/hip_bf16.h>

typedef __attribute__((ext_vector_type(16))) __bf16 v16bf;
typedef __attribute__((ext_vector_type(8)))  __bf16 v8bf;
typedef __attribute__((ext_vector_type(8)))  float  v8f;
typedef __attribute__((ext_vector_type(4)))  int    v4i;

#define O_DIM 4096
#define I_DIM 4096
#define M_DIM 8192   // B*S = 4*2048
#define BM 128
#define BN 128
#define BK 32
#define LDP 40       // padded LDS row length (elements): 32 + 8 -> 80B stride, conflict-free b128 reads
#define NSTAGE 3     // async pipeline depth

#define GLOBAL_AS __attribute__((address_space(1)))
#define LOCAL_AS  __attribute__((address_space(3)))

typedef GLOBAL_AS v4i* gv4i_p;
typedef LOCAL_AS  v4i* lv4i_p;

__device__ __forceinline__ unsigned short f2bf(float f) {
    unsigned int u = __float_as_uint(f);
    u += 0x7FFFu + ((u >> 16) & 1u);   // round-to-nearest-even
    return (unsigned short)(u >> 16);
}

// ---- CDNA5 async copy: GLOBAL -> LDS, 16 bytes per lane, tracked by ASYNCcnt ----
__device__ __forceinline__ void async_copy_b128(const unsigned short* g, unsigned short* l) {
#if __has_builtin(__builtin_amdgcn_global_load_async_to_lds_b128)
    __builtin_amdgcn_global_load_async_to_lds_b128(
        (gv4i_p)(const void*)g, (lv4i_p)(void*)l, 0, 0);
#else
    const unsigned loff = (unsigned)(unsigned long long)l;   // LDS aperture: low 32 bits = ds offset
    asm volatile("global_load_async_to_lds_b128 %0, %1, off"
                 :: "v"(loff), "v"((unsigned long long)g) : "memory");
#endif
}

template <int N>
__device__ __forceinline__ void wait_async() {
#if __has_builtin(__builtin_amdgcn_s_wait_asynccnt)
    __builtin_amdgcn_s_wait_asynccnt(N);
#else
    asm volatile("s_wait_asynccnt %0" :: "i"(N) : "memory");
#endif
}

// ---------- Kernel 1: x (f32) -> bf16, 8 elements/thread ----------
__global__ void convert_x_kernel(const float* __restrict__ x, unsigned short* __restrict__ xb) {
    const size_t t = (size_t)blockIdx.x * blockDim.x + threadIdx.x;  // one per 8 elements
    const float4* p = (const float4*)x + t * 2;
    float4 a = p[0];
    float4 b = p[1];
    union { unsigned short s[8]; uint4 v; } r;
    r.s[0] = f2bf(a.x); r.s[1] = f2bf(a.y); r.s[2] = f2bf(a.z); r.s[3] = f2bf(a.w);
    r.s[4] = f2bf(b.x); r.s[5] = f2bf(b.y); r.s[6] = f2bf(b.z); r.s[7] = f2bf(b.w);
    ((uint4*)xb)[t] = r.v;
}

// ---------- Kernel 2: reconstruct effective weight row-by-row -> bf16 (exact) ----------
__global__ void build_weight_kernel(const float* __restrict__ owr,
                                    const float* __restrict__ delta,
                                    const float* __restrict__ zp,
                                    const float* __restrict__ aux,
                                    const float* __restrict__ lA,   // [4, I]
                                    const float* __restrict__ lB,   // [O, 4]
                                    unsigned short* __restrict__ wb) {
    const int o = blockIdx.x;
    const float z  = zp[o];
    const float d  = delta[o];
    const float b0 = lB[o * 4 + 0];
    const float b1 = lB[o * 4 + 1];
    const float b2 = lB[o * 4 + 2];
    const float b3 = lB[o * 4 + 3];
    const size_t row = (size_t)o * I_DIM;
    for (int i = threadIdx.x; i < I_DIM; i += blockDim.x) {
        const float wint = owr[row + i] - z;
        const float ba = b0 * lA[i] + b1 * lA[I_DIM + i] + b2 * lA[2 * I_DIM + i] + b3 * lA[3 * I_DIM + i];
        const float lw = (aux[row + i] + ba) / (wint == 0.0f ? 1.0f : wint);
        const float wu = d + lw;
        const float sgn = (wu > 0.0f) ? 1.0f : ((wu < 0.0f) ? -1.0f : 0.0f);
        const float shift = rintf(log2f(fabsf(wu) + 1e-16f));
        const float w = sgn * exp2f(shift) * wint;   // = +/- 2^shift * int, exact in bf16
        wb[row + i] = f2bf(w);
    }
}

// ---------- Kernel 3: WMMA GEMM out[M,N] = Xb[M,K] * Wb[N,K]^T + bias ----------
__global__ void __launch_bounds__(256)
gemm_bf16_wmma_kernel(const unsigned short* __restrict__ Xb,
                      const unsigned short* __restrict__ Wb,
                      const float* __restrict__ bias,
                      float* __restrict__ out) {
    __shared__ unsigned short As[NSTAGE][BM * LDP];
    __shared__ unsigned short Bs[NSTAGE][BN * LDP];

    const int tid  = threadIdx.x;
    const int lane = tid & 31;
    const int wave = tid >> 5;
    const int mBase = blockIdx.y * BM;
    const int nBase = blockIdx.x * BN;

    // 8 waves: 4 (M) x 2 (N); each wave -> 32(M) x 64(N) = 2x4 frags of 16x16
    const int wMoff = (wave & 3) * 32;
    const int wNoff = (wave >> 2) * 64;

    const v8f vzero = {0.f, 0.f, 0.f, 0.f, 0.f, 0.f, 0.f, 0.f};
    v8f acc[2][4];
#pragma unroll
    for (int mi = 0; mi < 2; ++mi)
#pragma unroll
        for (int ni = 0; ni < 4; ++ni) acc[mi][ni] = vzero;

    // cooperative global->LDS via async DMA: thread t covers rows lr and lr+64, 8 bf16 (16B) at col lc
    const int lr = tid >> 2;        // 0..63
    const int lc = (tid & 3) * 8;   // 0,8,16,24
    const unsigned short* xp0 = Xb + (size_t)(mBase + lr) * I_DIM + lc;
    const unsigned short* xp1 = Xb + (size_t)(mBase + lr + 64) * I_DIM + lc;
    const unsigned short* wp0 = Wb + (size_t)(nBase + lr) * I_DIM + lc;
    const unsigned short* wp1 = Wb + (size_t)(nBase + lr + 64) * I_DIM + lc;
    const int lOff0 = lr * LDP + lc;
    const int lOff1 = (lr + 64) * LDP + lc;

    const int nK = I_DIM / BK;

    // prologue: fill stages 0 and 1 (8 async ops in flight)
#pragma unroll
    for (int s = 0; s < 2; ++s) {
        const int k0 = s * BK;
        async_copy_b128(xp0 + k0, &As[s][lOff0]);
        async_copy_b128(xp1 + k0, &As[s][lOff1]);
        async_copy_b128(wp0 + k0, &Bs[s][lOff0]);
        async_copy_b128(wp1 + k0, &Bs[s][lOff1]);
    }

    // fragment lane addressing (ISA 7.12.2, 16-bit layouts)
    const int fr = lane & 15;
    const int aK = (lane >> 4) * 8;    // A: lanes>=16 hold K 8..15 / 24..31
    const int bK = (lane >> 4) * 8;    // B: lanes>=16 hold K 8..15 / 24..31

    int cur = 0;           // stage holding block kb
    int pre = NSTAGE - 1;  // stage to fill with block kb+2
    for (int kb = 0; kb < nK; ++kb) {
        // in-order async completion: <=4 outstanding guarantees block kb's copies landed
        if (kb + 1 < nK) wait_async<4>(); else wait_async<0>();
        __syncthreads();   // all waves' stage-kb data visible; fences last iteration's readers

        if (kb + 2 < nK) {
            // refill the stage whose readers were fenced by the barrier above
            const int k0 = (kb + 2) * BK;
            async_copy_b128(xp0 + k0, &As[pre][lOff0]);
            async_copy_b128(xp1 + k0, &As[pre][lOff1]);
            async_copy_b128(wp0 + k0, &Bs[pre][lOff0]);
            async_copy_b128(wp1 + k0, &Bs[pre][lOff1]);
        }

        v16bf afrag[2], bfrag[4];
#pragma unroll
        for (int mi = 0; mi < 2; ++mi) {
            const unsigned short* p = &As[cur][(wMoff + mi * 16 + fr) * LDP];
            v8bf lo = *(const v8bf*)(p + aK);
            v8bf hi = *(const v8bf*)(p + aK + 16);
            afrag[mi] = __builtin_shufflevector(lo, hi, 0, 1, 2, 3, 4, 5, 6, 7,
                                                8, 9, 10, 11, 12, 13, 14, 15);
        }
#pragma unroll
        for (int ni = 0; ni < 4; ++ni) {
            const unsigned short* p = &Bs[cur][(wNoff + ni * 16 + fr) * LDP];
            v8bf lo = *(const v8bf*)(p + bK);
            v8bf hi = *(const v8bf*)(p + bK + 16);
            bfrag[ni] = __builtin_shufflevector(lo, hi, 0, 1, 2, 3, 4, 5, 6, 7,
                                                8, 9, 10, 11, 12, 13, 14, 15);
        }

#pragma unroll
        for (int mi = 0; mi < 2; ++mi)
#pragma unroll
            for (int ni = 0; ni < 4; ++ni)
                acc[mi][ni] = __builtin_amdgcn_wmma_f32_16x16x32_bf16(
                    false, afrag[mi], false, bfrag[ni], (short)0, acc[mi][ni], false, false);

        cur = (cur == NSTAGE - 1) ? 0 : cur + 1;
        pre = (pre == NSTAGE - 1) ? 0 : pre + 1;
    }

    // C/D layout: lane<16 -> N=lane, M=r ; lane>=16 -> N=lane-16, M=r+8
    const int ncol0 = nBase + wNoff + fr;
    const int mrow0 = mBase + wMoff + ((lane >> 4) * 8);
#pragma unroll
    for (int ni = 0; ni < 4; ++ni) {
        const int nn = ncol0 + ni * 16;
        const float bv = bias[nn];
#pragma unroll
        for (int mi = 0; mi < 2; ++mi) {
            const int m0 = mrow0 + mi * 16;
#pragma unroll
            for (int r = 0; r < 8; ++r)
                out[(size_t)(m0 + r) * O_DIM + nn] = acc[mi][ni][r] + bv;
        }
    }
}

extern "C" void kernel_launch(void* const* d_in, const int* in_sizes, int n_in,
                              void* d_out, int out_size, void* d_ws, size_t ws_size,
                              hipStream_t stream) {
    const float* x     = (const float*)d_in[0];
    const float* owr   = (const float*)d_in[1];
    const float* delta = (const float*)d_in[2];
    const float* zp    = (const float*)d_in[3];
    const float* aux   = (const float*)d_in[4];
    const float* lA    = (const float*)d_in[5];
    const float* lB    = (const float*)d_in[6];
    const float* bias  = (const float*)d_in[7];
    float* outp = (float*)d_out;

    unsigned short* Wb  = (unsigned short*)d_ws;                // 4096*4096*2 = 32 MB
    unsigned short* Xbf = Wb + (size_t)O_DIM * I_DIM;           // 8192*4096*2 = 64 MB

    // 1) x -> bf16
    {
        const size_t n8 = (size_t)M_DIM * I_DIM / 8;
        convert_x_kernel<<<dim3((unsigned)(n8 / 256)), dim3(256), 0, stream>>>(x, Xbf);
    }
    // 2) weight reconstruction -> bf16 (exact: +/- 2^s * int)
    build_weight_kernel<<<dim3(O_DIM), dim3(256), 0, stream>>>(owr, delta, zp, aux, lA, lB, Wb);
    // 3) WMMA GEMM + bias (3-stage async-LDS pipeline)
    {
        dim3 grid(O_DIM / BN, M_DIM / BM);   // 32 x 64 blocks
        gemm_bf16_wmma_kernel<<<grid, dim3(256), 0, stream>>>(Xbf, Wb, bias, outp);
    }
}
